// Attention_19868518711490
// MI455X (gfx1250) — matrix operand order repaired and checked by reading, not compiled, not run on hardware
//
#include <hip/hip_runtime.h>
#include <hip/hip_bf16.h>

// ---------------------------------------------------------------------------
// MHA block for MI455X (gfx1250, wave32, WMMA f32_16x16x32_f16)
//   qkv = x @ Wqkv + b ; per-head flash attention ; out = att @ Wproj + b
// f16 storage, f32 WMMA accumulation, register-blocked GEMMs (2x2 = 32x32
// per wave: ~90 VGPRs, no spills, 4 WMMAs per 8 b128 loads per K-step).
// ---------------------------------------------------------------------------

typedef _Float16 half_t;
typedef __attribute__((ext_vector_type(16))) _Float16 v16h;
typedef __attribute__((ext_vector_type(8)))  _Float16 v8h;
typedef __attribute__((ext_vector_type(8)))  float    v8f;

union V16U { v16h v; v8h h[2]; };

// A-operand (16x32, 16-bit): lane holds row (lane&15); element i -> K index
//   k(i) = i + 8*half + (i>=8 ? 8 : 0)   => chunks at [8*half, +16+8*half]
__device__ __forceinline__ v16h load_a16(const half_t* rowPtr, int kbase, int hl) {
  V16U u;
  u.h[0] = *(const v8h*)(rowPtr + kbase + 8 * hl);
  u.h[1] = *(const v8h*)(rowPtr + kbase + 16 + 8 * hl);
  return u.v;
}

// B-operand (32x16, 16-bit): lane holds column (lane&15); element i -> K index
//   k(i) = i + 8*half + (i>=8 ? 8 : 0)   => chunks at [8*half, +16+8*half]
__device__ __forceinline__ v16h load_b16(const half_t* rowPtr, int kbase, int hl) {
  V16U u;
  u.h[0] = *(const v8h*)(rowPtr + kbase + 8 * hl);
  u.h[1] = *(const v8h*)(rowPtr + kbase + 16 + 8 * hl);
  return u.v;
}

#define WMMA_F16(a, b, c) \
  __builtin_amdgcn_wmma_f32_16x16x32_f16(false, (a), false, (b), (short)0, (c), false, false)

// ------------------------- conversion kernels ------------------------------
__global__ void cvt_f32_f16(const float* __restrict__ src, half_t* __restrict__ dst, int n) {
  int i = blockIdx.x * blockDim.x + threadIdx.x;
  if (i < n) dst[i] = (half_t)src[i];
}

// dst[c][r] = (f16) src[r][c]   (src: rows x cols row-major)
__global__ void cvt_transpose_f16(const float* __restrict__ src, half_t* __restrict__ dst,
                                  int rows, int cols) {
  int i = blockIdx.x * blockDim.x + threadIdx.x;
  if (i < rows * cols) {
    int r = i / cols, c = i - r * cols;
    dst[c * rows + r] = (half_t)src[i];
  }
}

// ------------------------- qkv GEMM (32x32 per wave) -----------------------
// C[4096,3072] = xh[4096,1024] @ Wqkv ; scatter into q (scaled), k, vT per head
__global__ void gemm_qkv(const half_t* __restrict__ xh, const half_t* __restrict__ wT,
                         const float* __restrict__ bias,
                         half_t* __restrict__ q, half_t* __restrict__ k,
                         half_t* __restrict__ vT) {
  const int wave = (blockIdx.x * blockDim.x + threadIdx.x) >> 5;
  const int lane = threadIdx.x & 31;
  const int n = lane & 15, hl = lane >> 4;
  const int tm = wave / 96;           // 128 row blocks of 32 rows
  const int tc = wave - tm * 96;      // 96 col blocks of 32 cols

  const half_t* arow0 = xh + (tm * 32 + n) * 1024;
  const half_t* arow1 = arow0 + 16 * 1024;
  const half_t* brow0 = wT + (tc * 32 + n) * 1024;
  const half_t* brow1 = brow0 + 16 * 1024;

  v8f acc00 = {}, acc01 = {}, acc10 = {}, acc11 = {};
  for (int kb = 0; kb < 1024; kb += 32) {
    v16h a0 = load_a16(arow0, kb, hl);
    v16h a1 = load_a16(arow1, kb, hl);
    v16h b0 = load_b16(brow0, kb, hl);
    v16h b1 = load_b16(brow1, kb, hl);
    acc00 = WMMA_F16(a0, b0, acc00);
    acc01 = WMMA_F16(a0, b1, acc01);
    acc10 = WMMA_F16(a1, b0, acc10);
    acc11 = WMMA_F16(a1, b1, acc11);
  }

  // 32-wide col block sits inside one head (64) and one part (1024) -> uniform
  const int colbase = tc * 32;
  const int part = colbase >> 10;     // 0:q 1:k 2:v
  const int head = (colbase & 1023) >> 6;

  v8f accs[2][2] = {{acc00, acc01}, {acc10, acc11}};
#pragma unroll
  for (int cs = 0; cs < 2; ++cs) {
    const int c = colbase + cs * 16 + n;
    const float bi = bias[c];
    const int d = c & 63;
#pragma unroll
    for (int r = 0; r < 2; ++r) {
#pragma unroll
      for (int j = 0; j < 8; ++j) {
        const int token = tm * 32 + r * 16 + j + 8 * hl;  // C layout: M = j+8*(lane>=16)
        const int bt = token >> 10, nt = token & 1023;
        const int bh = bt * 16 + head;
        const float val = accs[r][cs][j] + bi;
        if (part == 0)      q [(bh * 1024 + nt) * 64 + d] = (half_t)(val * 0.125f);
        else if (part == 1) k [(bh * 1024 + nt) * 64 + d] = (half_t)val;
        else                vT[(bh * 64 + d) * 1024 + nt] = (half_t)val;  // V transposed
      }
    }
  }
}

// ------------------------- flash attention ---------------------------------
// One wave per (b, h, 16-query tile). Computes S^T = K.Q^T so that the softmaxed
// P tile is already in WMMA A-layout (no cross-lane transpose needed).
__global__ void attn_kernel(const half_t* __restrict__ q, const half_t* __restrict__ k,
                            const half_t* __restrict__ vT, half_t* __restrict__ att) {
  const int wave = (blockIdx.x * blockDim.x + threadIdx.x) >> 5;
  const int lane = threadIdx.x & 31;
  const int n = lane & 15, hl = lane >> 4;

  const int qt = wave & 63;           // query tile
  const int bh = wave >> 6;           // b*16 + h
  const int b  = bh >> 4;
  const int h  = bh & 15;

  const half_t* qp = q  + bh * (1024 * 64);
  const half_t* kp = k  + bh * (1024 * 64);
  const half_t* vp = vT + bh * (64 * 1024);

  // B operand = Q^T : lane's column = query row (loaded once, reused 32x)
  const half_t* qrow = qp + (qt * 16 + n) * 64;
  const v16h bq0 = load_b16(qrow, 0, hl);
  const v16h bq1 = load_b16(qrow, 32, hl);

  v8f o[4] = {v8f{}, v8f{}, v8f{}, v8f{}};
  float m_run = -1e30f;
  float l_run = 0.0f;

  for (int kb = 0; kb < 32; ++kb) {
    const int keybase = kb * 32;

    // A operands = K rows (keys as M). Two stacked 16-key tiles.
    const half_t* k0p = kp + (keybase + n) * 64;
    const half_t* k1p = kp + (keybase + 16 + n) * 64;
    v16h ak00 = load_a16(k0p, 0, hl),  ak01 = load_a16(k0p, 32, hl);
    v16h ak10 = load_a16(k1p, 0, hl),  ak11 = load_a16(k1p, 32, hl);

    v8f c0 = {}, c1 = {};
    c0 = WMMA_F16(ak00, bq0, c0);
    c0 = WMMA_F16(ak01, bq1, c0);
    c1 = WMMA_F16(ak10, bq0, c1);
    c1 = WMMA_F16(ak11, bq1, c1);

    // online softmax; this lane holds 16 of the 32 key scores for query n
    float bm = -1e30f;
#pragma unroll
    for (int j = 0; j < 8; ++j) { bm = fmaxf(bm, c0[j]); bm = fmaxf(bm, c1[j]); }
    bm = fmaxf(bm, __shfl_xor(bm, 16, 32));
    const float m_new = fmaxf(m_run, bm);
    const float fac = __expf(m_run - m_new);

    float e0[8], e1[8], rs = 0.0f;
#pragma unroll
    for (int j = 0; j < 8; ++j) {
      e0[j] = __expf(c0[j] - m_new);
      e1[j] = __expf(c1[j] - m_new);
      rs += e0[j] + e1[j];
    }
    rs += __shfl_xor(rs, 16, 32);
    l_run = l_run * fac + rs;
    m_run = m_new;

    // broadcast rescale factor to O-tile row positions (M = j + 8*hl)
    float facM[8];
#pragma unroll
    for (int j = 0; j < 8; ++j) facM[j] = __shfl(fac, j + 8 * hl, 32);

    // P: S^T C-layout == P A-layout (in-lane repack only)
    V16U pu;
#pragma unroll
    for (int i = 0; i < 8; ++i) { pu.v[i] = (_Float16)e0[i]; pu.v[i + 8] = (_Float16)e1[i]; }
    const v16h pa = pu.v;

    // O += P @ V  over 4 d-slices; B operand rows come from V^T (row-major)
#pragma unroll
    for (int t = 0; t < 4; ++t) {
      const v16h bv = load_b16(vp + (t * 16 + n) * 1024, keybase, hl);
      v8f oc;
#pragma unroll
      for (int j = 0; j < 8; ++j) oc[j] = o[t][j] * facM[j];
      o[t] = WMMA_F16(pa, bv, oc);
    }
  }

  float lM[8];
#pragma unroll
  for (int j = 0; j < 8; ++j) lM[j] = __shfl(l_run, j + 8 * hl, 32);

  half_t* outp = att + (b * 1024 + qt * 16) * 1024 + h * 64;
#pragma unroll
  for (int t = 0; t < 4; ++t)
#pragma unroll
    for (int j = 0; j < 8; ++j) {
      const int row = j + 8 * hl;
      outp[row * 1024 + t * 16 + n] = (half_t)(o[t][j] / lM[j]);
    }
}

// ------------------------- output projection (32x32 per wave) --------------
__global__ void gemm_proj(const half_t* __restrict__ att, const half_t* __restrict__ wT,
                          const float* __restrict__ bias, float* __restrict__ out) {
  const int wave = (blockIdx.x * blockDim.x + threadIdx.x) >> 5;
  const int lane = threadIdx.x & 31;
  const int n = lane & 15, hl = lane >> 4;
  const int tm = wave >> 5;           // 128 row blocks of 32 rows
  const int tc = wave & 31;           // 32 col blocks of 32 cols

  const half_t* arow0 = att + (tm * 32 + n) * 1024;
  const half_t* arow1 = arow0 + 16 * 1024;
  const half_t* brow0 = wT + (tc * 32 + n) * 1024;
  const half_t* brow1 = brow0 + 16 * 1024;

  v8f acc00 = {}, acc01 = {}, acc10 = {}, acc11 = {};
  for (int kb = 0; kb < 1024; kb += 32) {
    v16h a0 = load_a16(arow0, kb, hl);
    v16h a1 = load_a16(arow1, kb, hl);
    v16h b0 = load_b16(brow0, kb, hl);
    v16h b1 = load_b16(brow1, kb, hl);
    acc00 = WMMA_F16(a0, b0, acc00);
    acc01 = WMMA_F16(a0, b1, acc01);
    acc10 = WMMA_F16(a1, b0, acc10);
    acc11 = WMMA_F16(a1, b1, acc11);
  }

  v8f accs[2][2] = {{acc00, acc01}, {acc10, acc11}};
#pragma unroll
  for (int cs = 0; cs < 2; ++cs) {
    const int c = tc * 32 + cs * 16 + n;
    const float bi = bias[c];
#pragma unroll
    for (int r = 0; r < 2; ++r) {
#pragma unroll
      for (int j = 0; j < 8; ++j) {
        const int token = tm * 32 + r * 16 + j + 8 * hl;
        out[token * 1024 + c] = accs[r][cs][j] + bi;
      }
    }
  }
}

// ------------------------- launcher ----------------------------------------
extern "C" void kernel_launch(void* const* d_in, const int* in_sizes, int n_in,
                              void* d_out, int out_size, void* d_ws, size_t ws_size,
                              hipStream_t stream) {
  const float* x      = (const float*)d_in[0];   // [4,1024,1024]
  const float* w_qkv  = (const float*)d_in[1];   // [1024,3072]
  const float* b_qkv  = (const float*)d_in[2];   // [3072]
  const float* w_proj = (const float*)d_in[3];   // [1024,1024]
  const float* b_proj = (const float*)d_in[4];   // [1024]
  float* out = (float*)d_out;

  char* ws = (char*)d_ws;
  half_t* xh     = (half_t*)(ws + 0);           //  8 MB: x as f16
  half_t* wqkvT  = (half_t*)(ws + (8u  << 20)); //  6 MB: Wqkv^T f16
  half_t* wprojT = (half_t*)(ws + (14u << 20)); //  2 MB: Wproj^T f16
  half_t* Q      = (half_t*)(ws + (16u << 20)); //  8 MB: q scaled [b,h,n,d]
  half_t* K      = (half_t*)(ws + (24u << 20)); //  8 MB: k        [b,h,n,d]
  half_t* VT     = (half_t*)(ws + (32u << 20)); //  8 MB: v^T      [b,h,d,n]
  half_t* ATT    = (half_t*)(ws + (40u << 20)); //  8 MB: att out  [token,1024]

  cvt_f32_f16      <<<16384, 256, 0, stream>>>(x, xh, 4096 * 1024);
  cvt_transpose_f16<<<12288, 256, 0, stream>>>(w_qkv, wqkvT, 1024, 3072);
  cvt_transpose_f16<<< 4096, 256, 0, stream>>>(w_proj, wprojT, 1024, 1024);

  // 128x96 = 12288 waves, 8 waves/block
  gemm_qkv   <<<1536, 256, 0, stream>>>(xh, wqkvT, b_qkv, Q, K, VT);
  // 4*16*64 = 4096 waves
  attn_kernel<<< 512, 256, 0, stream>>>(Q, K, VT, ATT);
  // 128x32 = 4096 waves
  gemm_proj  <<< 512, 256, 0, stream>>>(ATT, wprojT, b_proj, out);
}
